// AttentionStack_53764400611883
// MI455X (gfx1250) — matrix operand order repaired and checked by reading, not compiled, not run on hardware
//
#include <hip/hip_runtime.h>

// ---------------- CDNA5 WMMA types ----------------
typedef _Float16 half_t;
typedef __attribute__((ext_vector_type(16))) _Float16 v16h;
typedef __attribute__((ext_vector_type(8)))  float    v8f;

#define WMMA_F32_F16(a, b, c) \
  __builtin_amdgcn_wmma_f32_16x16x32_f16(false, (a), false, (b), (short)0, (c), false, false)

// ---------------- problem constants ----------------
constexpr int B_   = 4;
constexpr int S_   = 1024;   // T*H*W
constexpr int C_   = 576;
constexpr int NH_  = 8;
constexpr int DK_  = 72;
constexpr int DKP_ = 96;     // DK padded to 3x32 for WMMA K-dim
constexpr int FF_  = 2304;   // 4*C
constexpr int M_   = B_ * S_;  // 4096 rows in all GEMMs
constexpr int L_   = 4;

enum { MODE_QKV = 0, MODE_RESID = 1, MODE_GELU = 2, MODE_VT = 3 };

// ---------------- fragment loaders (ISA 7.12.2 layouts, wave32) --------------
// All loaders fetch two 16-byte contiguous chunks per lane -> *_load_b128.

// A (16x32 f16) from row-major src: lane m = lane&15; k-runs split by half-wave.
// a[0..7] = src[m][hi*8 .. +7], a[8..15] = src[m][16+hi*8 .. +7].
__device__ inline v16h load_a_frag(const half_t* __restrict__ src, int ld) {
  const int lane = threadIdx.x & 31;
  const int m = lane & 15, hi = lane >> 4;
  const half_t* p = src + m * ld + hi * 8;
  v16h a;
  *reinterpret_cast<uint4*>(&a)     = *reinterpret_cast<const uint4*>(p);
  *(reinterpret_cast<uint4*>(&a) + 1) = *reinterpret_cast<const uint4*>(p + 16);
  return a;
}

// B (32x16 f16) where the k-run is contiguous in memory:
// element (k, n) = src[n*ld + k]; lane n = lane&15, b[0..7] = k hi*8..+7, b[8..15] = k 16+hi*8..+7.
// Works for K^T from row-major K[key][d], for V^T[d][key], and for the
// fragment-ready column-major weight tile in LDS.
__device__ inline v16h load_b_contig(const half_t* __restrict__ src, int ld) {
  const int lane = threadIdx.x & 31;
  const int n = lane & 15, hi = lane >> 4;
  const half_t* p = src + n * ld + hi * 8;
  v16h b;
  *reinterpret_cast<uint4*>(&b)     = *reinterpret_cast<const uint4*>(p);
  *(reinterpret_cast<uint4*>(&b) + 1) = *reinterpret_cast<const uint4*>(p + 16);
  return b;
}

// ---------------- preprocess: right-shift + factored pos-embed ----------------
__global__ void prep_kernel(const float* __restrict__ x, const float* __restrict__ p0,
                            const float* __restrict__ p1, const float* __restrict__ p2,
                            float* __restrict__ xf) {
  const int idx = blockIdx.x * blockDim.x + threadIdx.x;
  if (idx >= B_ * S_ * C_) return;
  const int c = idx % C_;
  const int s = (idx / C_) % S_;
  const int b = idx / (C_ * S_);
  const float v = (s > 0) ? x[((size_t)b * S_ + (s - 1)) * C_ + c] : 0.f;
  const int t = s >> 8, hh = (s >> 4) & 15, ww = s & 15;
  float pe;
  if (c < 192)      pe = p0[t * 192 + c];
  else if (c < 384) pe = p1[hh * 192 + (c - 192)];
  else              pe = p2[ww * 192 + (c - 384)];
  xf[idx] = v + pe;
}

__global__ void zero_kernel(float4* __restrict__ p, int n4) {
  const int i = blockIdx.x * blockDim.x + threadIdx.x;
  if (i < n4) p[i] = make_float4(0.f, 0.f, 0.f, 0.f);
}

// ---------------- LayerNorm (fp32 math, (1e-5 + rsqrt(var)) per reference) ----
__global__ __launch_bounds__(256) void ln_kernel(const float* __restrict__ xf,
                                                 const float* __restrict__ g,
                                                 const float* __restrict__ bta,
                                                 half_t* __restrict__ out) {
  __shared__ float red[256];
  const int r = blockIdx.x;
  const float* row = xf + (size_t)r * C_;
  const int t = threadIdx.x;
  const float v0 = row[t];
  const float v1 = row[t + 256];
  const float v2 = (t < 64) ? row[t + 512] : 0.f;

  red[t] = v0 + v1 + v2;
  __syncthreads();
  for (int off = 128; off > 0; off >>= 1) {
    if (t < off) red[t] += red[t + off];
    __syncthreads();
  }
  const float mu = red[0] * (1.f / C_);
  __syncthreads();
  const float d0 = v0 - mu, d1 = v1 - mu, d2 = v2 - mu;
  red[t] = d0 * d0 + d1 * d1 + ((t < 64) ? d2 * d2 : 0.f);
  __syncthreads();
  for (int off = 128; off > 0; off >>= 1) {
    if (t < off) red[t] += red[t + off];
    __syncthreads();
  }
  const float var = red[0] * (1.f / C_);
  const float rs = 1e-5f + rsqrtf(var);

  out[(size_t)r * C_ + t]       = (half_t)(d0 * rs * g[t] + bta[t]);
  out[(size_t)r * C_ + t + 256] = (half_t)(d1 * rs * g[t + 256] + bta[t + 256]);
  if (t < 64)
    out[(size_t)r * C_ + t + 512] = (half_t)(d2 * rs * g[t + 512] + bta[t + 512]);
}

// ---------------- WMMA GEMM: C[4096,N] = A[4096,K](f16) * W[K,N](f32->f16) ---
// block tile 128(M) x 64(N), K-step 32, 8 waves as 4x2, each wave 32x32.
// Weight tile staged column-major (fragment-ready), row stride 40 halves so
// both staging stores and fragment loads hit 16 distinct LDS banks.
__global__ __launch_bounds__(256) void gemm_kernel(
    const half_t* __restrict__ A, int lda,
    const float* __restrict__ W, int ldw,
    const float* __restrict__ bias,
    float* __restrict__ resid,      // MODE_RESID: in/out fp32 stream
    half_t* __restrict__ outh,      // MODE_QKV / MODE_VT / MODE_GELU destination
    float scale, int mode, int K) {
  __shared__ half_t At[128][32];
  __shared__ half_t Bcm[64][40];    // [n][k], padded stride
  const int m0 = blockIdx.y * 128;
  const int n0 = blockIdx.x * 64;
  const int N = (int)gridDim.x * 64;
  const int tid = threadIdx.x;
  const int wave = tid >> 5;
  const int wm = wave >> 1, wn = wave & 1;

  v8f acc[2][2] = {{{}, {}}, {{}, {}}};

  for (int kk = 0; kk < K; kk += 32) {
    // stage A tile (f16, 128x32) - 512 uint4 chunks / 256 threads
#pragma unroll
    for (int it = 0; it < 2; ++it) {
      const int id = tid + it * 256;
      const int row = id >> 2, ch = id & 3;
      const uint4* src = reinterpret_cast<const uint4*>(A + (size_t)(m0 + row) * lda + kk) + ch;
      *(reinterpret_cast<uint4*>(&At[row][0]) + ch) = *src;
    }
    // stage W tile (fp32 -> f16, 32x64) into column-major Bcm[n][k]
#pragma unroll
    for (int it = 0; it < 2; ++it) {
      const int id = tid + it * 256;
      const int row = id >> 4, c4 = id & 15;   // k = row, n = 4*c4..4*c4+3
      const float4 w4 = *(reinterpret_cast<const float4*>(W + (size_t)(kk + row) * ldw + n0) + c4);
      Bcm[c4 * 4 + 0][row] = (half_t)w4.x;
      Bcm[c4 * 4 + 1][row] = (half_t)w4.y;
      Bcm[c4 * 4 + 2][row] = (half_t)w4.z;
      Bcm[c4 * 4 + 3][row] = (half_t)w4.w;
    }
    if (kk + 32 < K)  // emits global_prefetch_b8 for next weight tile
      __builtin_prefetch(W + (size_t)(kk + 32) * ldw + n0, 0, 1);
    __syncthreads();

    const v16h a0 = load_a_frag(&At[wm * 32][0], 32);
    const v16h a1 = load_a_frag(&At[wm * 32 + 16][0], 32);
    const v16h b0 = load_b_contig(&Bcm[wn * 32][0], 40);
    const v16h b1 = load_b_contig(&Bcm[wn * 32 + 16][0], 40);
    acc[0][0] = WMMA_F32_F16(a0, b0, acc[0][0]);
    acc[0][1] = WMMA_F32_F16(a0, b1, acc[0][1]);
    acc[1][0] = WMMA_F32_F16(a1, b0, acc[1][0]);
    acc[1][1] = WMMA_F32_F16(a1, b1, acc[1][1]);
    __syncthreads();
  }

  // epilogue (C/D layout: m = vr + 8*(lane>=16), n = lane&15)
  const int lane = tid & 31;
  const int cn = lane & 15, hi = lane >> 4;
#pragma unroll
  for (int fm = 0; fm < 2; ++fm)
#pragma unroll
    for (int fn = 0; fn < 2; ++fn)
#pragma unroll
      for (int vr = 0; vr < 8; ++vr) {
        const int m = m0 + wm * 32 + fm * 16 + hi * 8 + vr;
        const int n = n0 + wn * 32 + fn * 16 + cn;
        const float val = acc[fm][fn][vr];
        if (mode == MODE_QKV) {                 // row-major padded [bh][s][96]
          const int b = m >> 10, s = m & 1023;
          const int head = n / DK_, d = n % DK_;
          outh[((size_t)(b * NH_ + head) * S_ + s) * DKP_ + d] = (half_t)(val * scale);
        } else if (mode == MODE_VT) {           // transposed padded [bh][96][S]
          const int b = m >> 10, s = m & 1023;
          const int head = n / DK_, d = n % DK_;
          outh[((size_t)(b * NH_ + head) * DKP_ + d) * S_ + s] = (half_t)val;
        } else if (mode == MODE_RESID) {
          const size_t o = (size_t)m * N + n;
          resid[o] = resid[o] + val + bias[n];
        } else { // MODE_GELU: sigmoid(1.702*h)*h
          const float h = val + bias[n];
          outh[(size_t)m * N + n] = (half_t)(h / (1.f + __expf(-1.702f * h)));
        }
      }
}

// ---------------- flash attention: 1 wave per 16-row Q tile, 32-key steps ----
__global__ __launch_bounds__(256) void attn_kernel(const half_t* __restrict__ q,
                                                   const half_t* __restrict__ k,
                                                   const half_t* __restrict__ vt,
                                                   half_t* __restrict__ out) {
  __shared__ half_t pbuf[8][16 * 32];  // per-wave P tile, A-fragment-ready
  const int wave = threadIdx.x >> 5;
  const int lane = threadIdx.x & 31;
  const int bh = blockIdx.x;                   // 0..31 = b*NH + head
  const int qt = blockIdx.y * 8 + wave;        // 0..63
  const int qbase = qt * 16;
  const half_t* qp  = q  + (size_t)bh * S_ * DKP_;
  const half_t* kp  = k  + (size_t)bh * S_ * DKP_;
  const half_t* vtp = vt + (size_t)bh * DKP_ * S_;   // [d][key]
  const int batch = bh >> 3, head = bh & 7;
  const int cn = lane & 15, hi = lane >> 4;

  v16h aq[3];
#pragma unroll
  for (int kk = 0; kk < 3; ++kk)
    aq[kk] = load_a_frag(qp + (size_t)qbase * DKP_ + kk * 32, DKP_);

  float mrow[8], lrow[8];
  v8f o[6] = {{}, {}, {}, {}, {}, {}};
#pragma unroll
  for (int i = 0; i < 8; ++i) { mrow[i] = -1e30f; lrow[i] = 0.f; }

  const int jsteps = (qbase + 15) / 32 + 1;    // causal column limit
  for (int j = 0; j < jsteps; ++j) {
    const int jb = j * 32;
    v8f s0 = {}, s1 = {};
#pragma unroll
    for (int kk = 0; kk < 3; ++kk) {           // K dim = 96 (padded with zeros)
      // B[k=d][n=key] = K[jb+n][d] : contiguous d-run per lane
      const v16h b0 = load_b_contig(kp + (size_t)jb * DKP_ + kk * 32, DKP_);
      const v16h b1 = load_b_contig(kp + (size_t)(jb + 16) * DKP_ + kk * 32, DKP_);
      s0 = WMMA_F32_F16(aq[kk], b0, s0);
      s1 = WMMA_F32_F16(aq[kk], b1, s1);
    }
    // causal mask + online softmax (row = qbase + hi*8 + vr, col = jb + cn [+16])
#pragma unroll
    for (int vr = 0; vr < 8; ++vr) {
      const int row = qbase + hi * 8 + vr;
      if (jb + cn > row)      s0[vr] = -1e30f;
      if (jb + 16 + cn > row) s1[vr] = -1e30f;
      float tmax = fmaxf(s0[vr], s1[vr]);
#pragma unroll
      for (int mk = 8; mk >= 1; mk >>= 1) tmax = fmaxf(tmax, __shfl_xor(tmax, mk, 32));
      const float mnew  = fmaxf(mrow[vr], tmax);
      const float alpha = __expf(mrow[vr] - mnew);
      const float p0 = __expf(s0[vr] - mnew);
      const float p1 = __expf(s1[vr] - mnew);
      float psum = p0 + p1;
#pragma unroll
      for (int mk = 8; mk >= 1; mk >>= 1) psum += __shfl_xor(psum, mk, 32);
      lrow[vr] = lrow[vr] * alpha + psum;
      mrow[vr] = mnew;
#pragma unroll
      for (int nt = 0; nt < 6; ++nt) o[nt][vr] *= alpha;
      const int m = hi * 8 + vr;
      pbuf[wave][m * 32 + cn]      = (half_t)p0;
      pbuf[wave][m * 32 + 16 + cn] = (half_t)p1;
    }
    asm volatile("s_wait_dscnt 0" ::: "memory");  // cross-lane LDS visibility
    const v16h ap = load_a_frag(&pbuf[wave][0], 32);
#pragma unroll
    for (int nt = 0; nt < 6; ++nt) {              // P(16x32) x V(32x96)
      // B[k=key][n=d] = Vt[d][jb+key] : contiguous key-run per lane
      const v16h bv = load_b_contig(vtp + (size_t)(nt * 16) * S_ + jb, S_);
      o[nt] = WMMA_F32_F16(ap, bv, o[nt]);
    }
  }

  // gather compact output [b, s, head*72 + d], drop the 72..95 pad columns
#pragma unroll
  for (int nt = 0; nt < 5; ++nt)
#pragma unroll
    for (int vr = 0; vr < 8; ++vr) {
      const int d = nt * 16 + cn;
      if (d < DK_) {
        const int srow = qbase + hi * 8 + vr;
        out[((size_t)(batch * S_ + srow)) * C_ + head * DK_ + d] =
            (half_t)(o[nt][vr] / lrow[vr]);
      }
    }
}

// ---------------- host launch ----------------
extern "C" void kernel_launch(void* const* d_in, const int* in_sizes, int n_in,
                              void* d_out, int out_size, void* d_ws, size_t ws_size,
                              hipStream_t stream) {
  (void)in_sizes; (void)n_in; (void)out_size; (void)ws_size;
  const float* x     = (const float*)d_in[0];
  const float* pos0  = (const float*)d_in[1];
  const float* pos1  = (const float*)d_in[2];
  const float* pos2  = (const float*)d_in[3];
  const float* ln1_g = (const float*)d_in[4];
  const float* ln1_b = (const float*)d_in[5];
  const float* wq    = (const float*)d_in[6];
  const float* wk    = (const float*)d_in[7];
  const float* wv    = (const float*)d_in[8];
  const float* wo    = (const float*)d_in[9];
  const float* wo_b  = (const float*)d_in[10];
  const float* ln2_g = (const float*)d_in[11];
  const float* ln2_b = (const float*)d_in[12];
  const float* w1    = (const float*)d_in[13];
  const float* b1    = (const float*)d_in[14];
  const float* w2    = (const float*)d_in[15];
  const float* b2    = (const float*)d_in[16];

  // workspace carve (all region sizes are multiples of 256B)
  char* ws = (char*)d_ws;
  float*  xf   = (float*)ws;  ws += (size_t)M_ * C_ * sizeof(float);     // 9.44 MB
  half_t* h16  = (half_t*)ws; ws += (size_t)M_ * C_ * sizeof(half_t);    // 4.72 MB
  half_t* qb   = (half_t*)ws; ws += (size_t)B_ * NH_ * S_ * DKP_ * 2;    // 6.29 MB
  half_t* kb   = (half_t*)ws; ws += (size_t)B_ * NH_ * S_ * DKP_ * 2;
  half_t* vtb  = (half_t*)ws; ws += (size_t)B_ * NH_ * DKP_ * S_ * 2;    // V^T
  half_t* aout = (half_t*)ws; ws += (size_t)M_ * C_ * sizeof(half_t);    // 4.72 MB
  half_t* hid  = (half_t*)ws; ws += (size_t)M_ * FF_ * sizeof(half_t);   // 18.9 MB

  // zero the padded q/k/vt buffers (contiguous) so pads contribute 0
  {
    const int n4 = (int)((3ull * B_ * NH_ * S_ * DKP_ * 2) / 16);
    zero_kernel<<<(n4 + 255) / 256, 256, 0, stream>>>((float4*)qb, n4);
  }
  prep_kernel<<<(M_ * C_ + 255) / 256, 256, 0, stream>>>(x, pos0, pos1, pos2, xf);

  const float qscale = 0.11785113019775792f;  // 1/sqrt(72)
  const dim3 gC(C_ / 64, M_ / 128);    // (9, 32)  N=576 GEMMs
  const dim3 gF(FF_ / 64, M_ / 128);   // (36, 32) N=2304 GEMM

  for (int l = 0; l < L_; ++l) {
    const size_t wCC = (size_t)l * C_ * C_;
    const size_t wCF = (size_t)l * C_ * FF_;
    // LN1 -> h16
    ln_kernel<<<M_, 256, 0, stream>>>(xf, ln1_g + l * C_, ln1_b + l * C_, h16);
    // QKV projections (scale folded into Q, V stored transposed)
    gemm_kernel<<<gC, 256, 0, stream>>>(h16, C_, wq + wCC, C_, nullptr, nullptr, qb,
                                        qscale, MODE_QKV, C_);
    gemm_kernel<<<gC, 256, 0, stream>>>(h16, C_, wk + wCC, C_, nullptr, nullptr, kb,
                                        1.f, MODE_QKV, C_);
    gemm_kernel<<<gC, 256, 0, stream>>>(h16, C_, wv + wCC, C_, nullptr, nullptr, vtb,
                                        1.f, MODE_VT, C_);
    // causal flash attention
    attn_kernel<<<dim3(B_ * NH_, S_ / 16 / 8), 256, 0, stream>>>(qb, kb, vtb, aout);
    // output projection + bias + residual
    gemm_kernel<<<gC, 256, 0, stream>>>(aout, C_, wo + wCC, C_, wo_b + l * C_, xf,
                                        nullptr, 1.f, MODE_RESID, C_);
    // LN2 -> h16
    ln_kernel<<<M_, 256, 0, stream>>>(xf, ln2_g + l * C_, ln2_b + l * C_, h16);
    // FFN1 + bias + GeLU2 -> hid (f16)
    gemm_kernel<<<gF, 256, 0, stream>>>(h16, C_, w1 + wCF, FF_, b1 + (size_t)l * FF_,
                                        nullptr, hid, 1.f, MODE_GELU, C_);
    // FFN2 + bias + residual
    gemm_kernel<<<gC, 256, 0, stream>>>(hid, FF_, w2 + wCF, C_, b2 + l * C_, xf,
                                        nullptr, 1.f, MODE_RESID, FF_);
  }

  hipMemcpyAsync(d_out, xf, (size_t)M_ * C_ * sizeof(float),
                 hipMemcpyDeviceToDevice, stream);
}
